// TransformerBlock_15040975471109
// MI455X (gfx1250) — compile-verified, matrix operand order checked by reading, not run on hardware
//
#include <hip/hip_runtime.h>
#include <hip/hip_bf16.h>
#include <stdint.h>

// ---------------------------------------------------------------------------
// MI455X (gfx1250, wave32) transformer block:
//   y   = x + Attn(RMSNorm(x,g1)) @ Wo^T
//   out = y + SwiGLU(RMSNorm(y,g2)) @ W2^T
// All matmuls run on v_wmma_f32_16x16x32_bf16; big GEMMs are LDS-staged and
// double-buffered, staging via global_load_async_to_lds_b128 when available.
// ---------------------------------------------------------------------------

typedef __bf16 bf16_t;
typedef __bf16 bf16x16 __attribute__((ext_vector_type(16)));
typedef __bf16 bf16x8  __attribute__((ext_vector_type(8)));
typedef float  f32x8   __attribute__((ext_vector_type(8)));
typedef int    v4i     __attribute__((ext_vector_type(4)));

#define D_MODEL 1024
#define D_FF    4096
#define NHEAD   16
#define DK      64
#define BATCH   4
#define SEQ     2048
#define MROWS   (BATCH * SEQ)   // 8192

#if __has_builtin(__builtin_amdgcn_global_load_async_to_lds_b128)
#define USE_ASYNC_LDS 1
#else
#define USE_ASYNC_LDS 0
#endif

__device__ __forceinline__ f32x8 wmma_bf16(bf16x16 a, bf16x16 b, f32x8 c) {
  return __builtin_amdgcn_wmma_f32_16x16x32_bf16(
      /*neg_a=*/false, a, /*neg_b=*/false, b,
      /*c_mod=*/(short)0, c, /*reuse_a=*/false, /*reuse_b=*/false);
}

// 16x32 (rows x K) bf16 fragment: lanes 0-15 hold K=k0..k0+15 of row
// (row0+lane), lanes 16-31 hold K=k0+16..k0+31.
__device__ __forceinline__ bf16x16 load_frag(const bf16_t* __restrict__ p,
                                             int ld, int row0, int k0) {
  const int lane = threadIdx.x & 31;
  const bf16_t* src = p + (size_t)(row0 + (lane & 15)) * ld
                        + (size_t)(k0 + ((lane >> 4) << 4));
  return *(const bf16x16*)src;
}

// ---------------------------------------------------------------------------
// LDS staging helpers (async-to-LDS when the builtin exists)
// ---------------------------------------------------------------------------
// Copies `elems` bf16 (tile row-major [rows][32]) from global (leading dim
// `ld`, origin row0/k0) into contiguous LDS, 16B per thread-step.
__device__ __forceinline__ void stage_tile(const bf16_t* __restrict__ g, int ld,
                                           int row0, int k0,
                                           bf16_t* s, int elems, int tid) {
  for (int f = tid * 8; f < elems; f += 256 * 8) {
    const int row = f >> 5;       // BK == 32
    const int col = f & 31;
    const bf16_t* gp = g + (size_t)(row0 + row) * ld + (size_t)(k0 + col);
#if USE_ASYNC_LDS
    __builtin_amdgcn_global_load_async_to_lds_b128(
        (v4i*)gp, (v4i*)(s + f), 0, 0);
#else
    *(bf16x8*)(s + f) = *(const bf16x8*)gp;
#endif
  }
}

__device__ __forceinline__ void stage_commit() {
#if USE_ASYNC_LDS
#if __has_builtin(__builtin_amdgcn_s_wait_asynccnt)
  __builtin_amdgcn_s_wait_asynccnt(0);
#else
  asm volatile("s_wait_asynccnt 0" ::: "memory");
#endif
#endif
  __syncthreads();
}

// ---------------------------------------------------------------------------
// fp32 -> bf16 elementwise conversion (weights)
// ---------------------------------------------------------------------------
__global__ void f32_to_bf16(const float* __restrict__ in,
                            bf16_t* __restrict__ out, int n) {
  int i = blockIdx.x * blockDim.x + threadIdx.x;
  if (i < n) out[i] = (bf16_t)in[i];
}

// ---------------------------------------------------------------------------
// RMSNorm (one 256-thread block per row) -> bf16 output
// ---------------------------------------------------------------------------
__global__ __launch_bounds__(256) void rmsnorm_bf16(const float* __restrict__ x,
                                                    const float* __restrict__ g,
                                                    bf16_t* __restrict__ out) {
  __shared__ float red[8];
  const size_t row = blockIdx.x;
  const float* xr = x + row * D_MODEL;
  float ss = 0.f;
  for (int i = threadIdx.x; i < D_MODEL; i += 256) {
    float v = xr[i];
    ss += v * v;
  }
#pragma unroll
  for (int off = 16; off > 0; off >>= 1) ss += __shfl_xor(ss, off, 32);
  if ((threadIdx.x & 31) == 0) red[threadIdx.x >> 5] = ss;
  __syncthreads();
  float tot = 0.f;
#pragma unroll
  for (int w = 0; w < 8; ++w) tot += red[w];
  const float rinv = rsqrtf(tot * (1.0f / D_MODEL) + 1e-5f);
  for (int i = threadIdx.x; i < D_MODEL; i += 256)
    out[row * D_MODEL + i] = (bf16_t)(xr[i] * rinv * g[i]);
}

// ---------------------------------------------------------------------------
// Fused QKV projection + RoPE with register double-buffering.
// One wave -> one 16x16 tile of each of Q,K,V.
// Writes q,k as bf16 [B,H,S,DK] (RoPE applied) and v as bf16 [B,H,DK,S].
// ---------------------------------------------------------------------------
__global__ __launch_bounds__(256) void qkv_rope(const bf16_t* __restrict__ xn,
                                                const bf16_t* __restrict__ wq,
                                                const bf16_t* __restrict__ wk,
                                                const bf16_t* __restrict__ wv,
                                                bf16_t* __restrict__ qo,
                                                bf16_t* __restrict__ ko,
                                                bf16_t* __restrict__ vto) {
  const int wave = blockIdx.x * 8 + (threadIdx.x >> 5);
  const int ntn = D_MODEL >> 4;        // 64 N-tiles
  const int tm = wave / ntn;
  const int tn = wave % ntn;
  const int lane = threadIdx.x & 31;
  const int half = lane >> 4;
  const int lx = lane & 15;

  const f32x8 zero = {0.f, 0.f, 0.f, 0.f, 0.f, 0.f, 0.f, 0.f};
  f32x8 aq = zero, ak = zero, av = zero;

  bf16x16 a  = load_frag(xn, D_MODEL, tm * 16, 0);
  bf16x16 bq = load_frag(wq, D_MODEL, tn * 16, 0);
  bf16x16 bk = load_frag(wk, D_MODEL, tn * 16, 0);
  bf16x16 bv = load_frag(wv, D_MODEL, tn * 16, 0);

  for (int kk = 0; kk < D_MODEL; kk += 32) {
    const int kn = (kk + 32 < D_MODEL) ? (kk + 32) : kk;
    bf16x16 a2 = load_frag(xn, D_MODEL, tm * 16, kn);
    bf16x16 q2 = load_frag(wq, D_MODEL, tn * 16, kn);
    bf16x16 k2 = load_frag(wk, D_MODEL, tn * 16, kn);
    bf16x16 v2 = load_frag(wv, D_MODEL, tn * 16, kn);
    __builtin_prefetch(xn + (size_t)(tm * 16) * D_MODEL + kn + 64, 0, 0);
    aq = wmma_bf16(a, bq, aq);
    ak = wmma_bf16(a, bk, ak);
    av = wmma_bf16(a, bv, av);
    a = a2; bq = q2; bk = k2; bv = v2;
  }

  const int col = tn * 16 + lx;            // 0..1023
  const int h = col >> 6;
  const int d = col & 63;
  const int pair = d >> 1;                 // 0..31
  const bool even = (d & 1) == 0;
  const float inv_freq = __powf(10000.f, -(float)(2 * pair) * (1.0f / DK));

#pragma unroll
  for (int i = 0; i < 8; ++i) {
    const int m = tm * 16 + i + half * 8;  // global row = b*S + s
    const int s_pos = m & (SEQ - 1);
    const int bb = m >> 11;                // m / SEQ
    const float ang = (float)s_pos * inv_freq;
    const float c = __cosf(ang), sn = __sinf(ang);

    float vq = aq[i];
    float oq = __shfl_xor(vq, 1, 32);
    float x1 = even ? vq : oq, x2 = even ? oq : vq;
    float rq = even ? (x1 * c - x2 * sn) : (x1 * sn + x2 * c);

    float vk = ak[i];
    float ok = __shfl_xor(vk, 1, 32);
    float y1 = even ? vk : ok, y2 = even ? ok : vk;
    float rk = even ? (y1 * c - y2 * sn) : (y1 * sn + y2 * c);

    const size_t base = ((size_t)(bb * NHEAD + h) * SEQ + s_pos) * DK + d;
    qo[base] = (bf16_t)rq;
    ko[base] = (bf16_t)rk;
    vto[((size_t)(bb * NHEAD + h) * DK + d) * SEQ + s_pos] = (bf16_t)av[i];
  }
}

// ---------------------------------------------------------------------------
// Flash attention: one wave owns a 16-row query tile, streams causal K/V in
// j-blocks of 32.  Scores -> 4 WMMA, P re-laid-out through per-wave LDS,
// PV -> 4 WMMA.  Output written as bf16 ctx [M, D_MODEL].
// ---------------------------------------------------------------------------
#define ATTN_WAVES 4
__global__ __launch_bounds__(128) void attn_fa(const bf16_t* __restrict__ q,
                                               const bf16_t* __restrict__ k,
                                               const bf16_t* __restrict__ vT,
                                               bf16_t* __restrict__ ctx) {
  __shared__ __align__(64) bf16_t pbuf[ATTN_WAVES][16 * 32];
  const int wic = threadIdx.x >> 5;
  const int gwave = blockIdx.x * ATTN_WAVES + wic;
  const int nqt = SEQ >> 4;                 // 128
  const int qt = gwave % nqt;
  const int bh = gwave / nqt;               // 0..63
  const int lane = threadIdx.x & 31;
  const int half = lane >> 4;
  const int lx = lane & 15;

  const bf16_t* qb = q + (size_t)bh * SEQ * DK;
  const bf16_t* kb = k + (size_t)bh * SEQ * DK;
  const bf16_t* vb = vT + (size_t)bh * DK * SEQ;

  const bf16x16 qa0 = load_frag(qb, DK, qt * 16, 0);
  const bf16x16 qa1 = load_frag(qb, DK, qt * 16, 32);

  const f32x8 zero = {0.f, 0.f, 0.f, 0.f, 0.f, 0.f, 0.f, 0.f};
  f32x8 acc0 = zero, acc1 = zero, acc2 = zero, acc3 = zero;
  float mrow[8], lrow[8];
#pragma unroll
  for (int i = 0; i < 8; ++i) { mrow[i] = -1e30f; lrow[i] = 0.f; }

  const float scale = 0.125f;               // 1/sqrt(64)
  const int qend = qt * 16 + 15;

  for (int j0 = 0; j0 <= qend; j0 += 32) {
    f32x8 s0 = wmma_bf16(qa1, load_frag(kb, DK, j0, 32),
                         wmma_bf16(qa0, load_frag(kb, DK, j0, 0), zero));
    f32x8 s1 = wmma_bf16(qa1, load_frag(kb, DK, j0 + 16, 32),
                         wmma_bf16(qa0, load_frag(kb, DK, j0 + 16, 0), zero));

#pragma unroll
    for (int i = 0; i < 8; ++i) {
      const int qrow = qt * 16 + i + half * 8;
      s0[i] = ((j0 + lx) <= qrow) ? s0[i] * scale : -1e30f;
      s1[i] = ((j0 + 16 + lx) <= qrow) ? s1[i] * scale : -1e30f;

      float mloc = fmaxf(s0[i], s1[i]);
#pragma unroll
      for (int off = 1; off < 16; off <<= 1)
        mloc = fmaxf(mloc, __shfl_xor(mloc, off, 32));
      const float mnew = fmaxf(mrow[i], mloc);
      const float corr = __expf(mrow[i] - mnew);
      const float p0 = __expf(s0[i] - mnew);
      const float p1 = __expf(s1[i] - mnew);
      float rs = p0 + p1;
#pragma unroll
      for (int off = 1; off < 16; off <<= 1) rs += __shfl_xor(rs, off, 32);
      lrow[i] = lrow[i] * corr + rs;
      mrow[i] = mnew;
      acc0[i] *= corr; acc1[i] *= corr; acc2[i] *= corr; acc3[i] *= corr;
      pbuf[wic][(i + half * 8) * 32 + lx] = (bf16_t)p0;
      pbuf[wic][(i + half * 8) * 32 + 16 + lx] = (bf16_t)p1;
    }

    // P as A-fragment (K = 32 key positions); wave-private LDS, DS ops in-order
    const bf16x16 pa = *(const bf16x16*)&pbuf[wic][lx * 32 + half * 16];
    acc0 = wmma_bf16(pa, load_frag(vb, SEQ, 0, j0), acc0);
    acc1 = wmma_bf16(pa, load_frag(vb, SEQ, 16, j0), acc1);
    acc2 = wmma_bf16(pa, load_frag(vb, SEQ, 32, j0), acc2);
    acc3 = wmma_bf16(pa, load_frag(vb, SEQ, 48, j0), acc3);
  }

  const int bb = bh >> 4;
  const int hh = bh & 15;
  const f32x8 accs[4] = {acc0, acc1, acc2, acc3};
#pragma unroll
  for (int t = 0; t < 4; ++t) {
#pragma unroll
    for (int i = 0; i < 8; ++i) {
      const int row = qt * 16 + i + half * 8;
      const int d = t * 16 + lx;
      ctx[((size_t)(bb * SEQ + row)) * D_MODEL + hh * DK + d] =
          (bf16_t)(accs[t][i] / lrow[i]);
    }
  }
}

// ---------------------------------------------------------------------------
// LDS-staged double-buffered GEMM: C = resid + A @ W^T
// Block = 256 thr (8 waves, 2Mx4N), tile BM=32 x BN=256, BK=32.
// LDS: 2*(2KB A + 16KB W) = 36KB.
// ---------------------------------------------------------------------------
__global__ __launch_bounds__(256) void gemm_resid_staged(
    const bf16_t* __restrict__ A, const bf16_t* __restrict__ W,
    const float* __restrict__ resid, float* __restrict__ C,
    int M, int N, int K) {
  __shared__ __align__(16) bf16_t sA[2][32 * 32];
  __shared__ __align__(16) bf16_t sW[2][256 * 32];

  const int tid = threadIdx.x;
  const int wid = tid >> 5;
  const int lane = tid & 31;
  const int half = lane >> 4;
  const int lx = lane & 15;
  const int nbn = N >> 8;                   // N / 256
  const int bm = blockIdx.x / nbn;
  const int bn = blockIdx.x % nbn;
  const int rowA0 = bm * 32;
  const int colB0 = bn * 256;
  const int wm = wid >> 2;                  // 0..1
  const int wn = wid & 3;                   // 0..3

  const f32x8 zero = {0.f, 0.f, 0.f, 0.f, 0.f, 0.f, 0.f, 0.f};
  f32x8 acc[4] = {zero, zero, zero, zero};

  stage_tile(A, K, rowA0, 0, sA[0], 32 * 32, tid);
  stage_tile(W, K, colB0, 0, sW[0], 256 * 32, tid);
  stage_commit();

  const int nk = K >> 5;
  int buf = 0;
  for (int ks = 0; ks < nk; ++ks) {
    if (ks + 1 < nk) {
      stage_tile(A, K, rowA0, (ks + 1) << 5, sA[buf ^ 1], 32 * 32, tid);
      stage_tile(W, K, colB0, (ks + 1) << 5, sW[buf ^ 1], 256 * 32, tid);
    }
    const bf16x16 a = load_frag(sA[buf], 32, wm * 16, 0);
#pragma unroll
    for (int t = 0; t < 4; ++t)
      acc[t] = wmma_bf16(a, load_frag(sW[buf], 32, wn * 64 + t * 16, 0), acc[t]);
    stage_commit();
    buf ^= 1;
  }

#pragma unroll
  for (int t = 0; t < 4; ++t) {
    const int n = colB0 + wn * 64 + t * 16 + lx;
#pragma unroll
    for (int i = 0; i < 8; ++i) {
      const int m = rowA0 + wm * 16 + i + half * 8;
      const size_t idx = (size_t)m * N + n;
      C[idx] = resid[idx] + acc[t][i];
    }
  }
}

// ---------------------------------------------------------------------------
// LDS-staged double-buffered SwiGLU up-projection:
//   h = silu(yn @ W1^T) * (yn @ W3^T), bf16 out [M, D_FF].
// Tile BM=32 x BN=128, BK=32.  LDS: 2*(2KB + 8KB + 8KB) = 36KB.
// ---------------------------------------------------------------------------
__global__ __launch_bounds__(256) void ffn_swiglu_staged(
    const bf16_t* __restrict__ yn, const bf16_t* __restrict__ w1,
    const bf16_t* __restrict__ w3, bf16_t* __restrict__ h,
    int M, int N, int K) {
  __shared__ __align__(16) bf16_t sA[2][32 * 32];
  __shared__ __align__(16) bf16_t sW1[2][128 * 32];
  __shared__ __align__(16) bf16_t sW3[2][128 * 32];

  const int tid = threadIdx.x;
  const int wid = tid >> 5;
  const int lane = tid & 31;
  const int half = lane >> 4;
  const int lx = lane & 15;
  const int nbn = N >> 7;                   // N / 128
  const int bm = blockIdx.x / nbn;
  const int bn = blockIdx.x % nbn;
  const int rowA0 = bm * 32;
  const int colB0 = bn * 128;
  const int wm = wid >> 2;                  // 0..1
  const int wn = wid & 3;                   // 0..3

  const f32x8 zero = {0.f, 0.f, 0.f, 0.f, 0.f, 0.f, 0.f, 0.f};
  f32x8 acc1[2] = {zero, zero};
  f32x8 acc3[2] = {zero, zero};

  stage_tile(yn, K, rowA0, 0, sA[0], 32 * 32, tid);
  stage_tile(w1, K, colB0, 0, sW1[0], 128 * 32, tid);
  stage_tile(w3, K, colB0, 0, sW3[0], 128 * 32, tid);
  stage_commit();

  const int nk = K >> 5;
  int buf = 0;
  for (int ks = 0; ks < nk; ++ks) {
    if (ks + 1 < nk) {
      stage_tile(yn, K, rowA0, (ks + 1) << 5, sA[buf ^ 1], 32 * 32, tid);
      stage_tile(w1, K, colB0, (ks + 1) << 5, sW1[buf ^ 1], 128 * 32, tid);
      stage_tile(w3, K, colB0, (ks + 1) << 5, sW3[buf ^ 1], 128 * 32, tid);
    }
    const bf16x16 a = load_frag(sA[buf], 32, wm * 16, 0);
#pragma unroll
    for (int t = 0; t < 2; ++t) {
      acc1[t] = wmma_bf16(a, load_frag(sW1[buf], 32, wn * 32 + t * 16, 0), acc1[t]);
      acc3[t] = wmma_bf16(a, load_frag(sW3[buf], 32, wn * 32 + t * 16, 0), acc3[t]);
    }
    stage_commit();
    buf ^= 1;
  }

#pragma unroll
  for (int t = 0; t < 2; ++t) {
    const int n = colB0 + wn * 32 + t * 16 + lx;
#pragma unroll
    for (int i = 0; i < 8; ++i) {
      const int m = rowA0 + wm * 16 + i + half * 8;
      const float av = acc1[t][i];
      const float uv = acc3[t][i];
      const float sw = av / (1.f + __expf(-av)) * uv;
      h[(size_t)m * N + n] = (bf16_t)sw;
    }
  }
}

// ---------------------------------------------------------------------------
// Host orchestration
// ---------------------------------------------------------------------------
extern "C" void kernel_launch(void* const* d_in, const int* in_sizes, int n_in,
                              void* d_out, int out_size, void* d_ws, size_t ws_size,
                              hipStream_t stream) {
  (void)in_sizes; (void)n_in; (void)out_size; (void)ws_size;

  const float* x  = (const float*)d_in[0];
  const float* wq = (const float*)d_in[1];
  const float* wk = (const float*)d_in[2];
  const float* wv = (const float*)d_in[3];
  const float* wo = (const float*)d_in[4];
  const float* g1 = (const float*)d_in[5];
  const float* g2 = (const float*)d_in[6];
  const float* w1 = (const float*)d_in[7];
  const float* w2 = (const float*)d_in[8];
  const float* w3 = (const float*)d_in[9];
  float* out = (float*)d_out;

  char* ws = (char*)d_ws;
  size_t off = 0;
  auto alloc = [&](size_t bytes) -> void* {
    void* p = ws + off;
    off = (off + bytes + 255) & ~(size_t)255;
    return p;
  };

  const size_t MD = (size_t)MROWS * D_MODEL;
  const size_t MF = (size_t)MROWS * D_FF;
  const size_t DD = (size_t)D_MODEL * D_MODEL;
  const size_t FD = (size_t)D_FF * D_MODEL;

  bf16_t* wqb = (bf16_t*)alloc(DD * 2);
  bf16_t* wkb = (bf16_t*)alloc(DD * 2);
  bf16_t* wvb = (bf16_t*)alloc(DD * 2);
  bf16_t* wob = (bf16_t*)alloc(DD * 2);
  bf16_t* w1b = (bf16_t*)alloc(FD * 2);
  bf16_t* w2b = (bf16_t*)alloc(FD * 2);
  bf16_t* w3b = (bf16_t*)alloc(FD * 2);
  bf16_t* xnb = (bf16_t*)alloc(MD * 2);
  bf16_t* qb  = (bf16_t*)alloc(MD * 2);
  bf16_t* kb  = (bf16_t*)alloc(MD * 2);
  bf16_t* vtb = (bf16_t*)alloc(MD * 2);
  bf16_t* ctx = (bf16_t*)alloc(MD * 2);
  float*  y   = (float*)alloc(MD * 4);
  bf16_t* ynb = (bf16_t*)alloc(MD * 2);
  bf16_t* hb  = (bf16_t*)alloc(MF * 2);

  // 1) weight conversion fp32 -> bf16
  auto cvt = [&](const float* src, bf16_t* dst, size_t n) {
    f32_to_bf16<<<(unsigned)((n + 255) / 256), 256, 0, stream>>>(src, dst, (int)n);
  };
  cvt(wq, wqb, DD); cvt(wk, wkb, DD); cvt(wv, wvb, DD); cvt(wo, wob, DD);
  cvt(w1, w1b, FD); cvt(w2, w2b, FD); cvt(w3, w3b, FD);

  // 2) RMSNorm(x, g1) -> xn (bf16)
  rmsnorm_bf16<<<MROWS, 256, 0, stream>>>(x, g1, xnb);

  // 3) fused QKV + RoPE: 512*64 waves -> 4096 blocks of 8 waves
  qkv_rope<<<4096, 256, 0, stream>>>(xnb, wqb, wkb, wvb, qb, kb, vtb);

  // 4) flash attention: 64 (b,h) * 128 q-tiles = 8192 waves -> 2048 blocks
  attn_fa<<<2048, 128, 0, stream>>>(qb, kb, vtb, ctx);

  // 5) y = x + ctx @ Wo^T : (8192/32)*(1024/256) = 1024 blocks
  gemm_resid_staged<<<1024, 256, 0, stream>>>(ctx, wob, x, y,
                                              MROWS, D_MODEL, D_MODEL);

  // 6) RMSNorm(y, g2) -> yn (bf16)
  rmsnorm_bf16<<<MROWS, 256, 0, stream>>>(y, g2, ynb);

  // 7) h = silu(yn@W1^T)*(yn@W3^T) : (8192/32)*(4096/128) = 8192 blocks
  ffn_swiglu_staged<<<8192, 256, 0, stream>>>(ynb, w1b, w3b, hb,
                                              MROWS, D_FF, D_MODEL);

  // 8) out = y + h @ W2^T : (8192/32)*(1024/256) = 1024 blocks
  gemm_resid_staged<<<1024, 256, 0, stream>>>(hb, w2b, y, out,
                                              MROWS, D_MODEL, D_FF);
}
